// CoOccurrenceMatrixFast_69157563400789
// MI455X (gfx1250) — compile-verified, matrix operand order checked by reading, not run on hardware
//
#include <hip/hip_runtime.h>
#include <hip/hip_bf16.h>
#include <stdint.h>

// ---------------------------------------------------------------------------
// CDNA5 (gfx1250): GLCM -> 6x conv (bf16 WMMA implicit GEMM, async-LDS
// staging) -> maxpools -> 3x FC (bf16 WMMA GEMM) -> sigmoid.
// bf16 over f16: GLCM bin counts (up to ~2.6e5) overflow f16's 65504 max.
// ---------------------------------------------------------------------------

typedef __bf16 bf16_t;
typedef __attribute__((address_space(3))) __bf16 lds_bf16;
typedef __attribute__((ext_vector_type(16))) __bf16 v16bf;
typedef __attribute__((ext_vector_type(8)))  float  v8f;

__device__ __forceinline__ bf16_t f32_to_bf16(float f) {
    union U { float f; uint32_t u; };
    U c; c.f = f;
    uint32_t u = c.u + 0x7FFFu + ((c.u >> 16) & 1u);   // round-to-nearest-even
    uint16_t h = (uint16_t)(u >> 16);
    bf16_t r; __builtin_memcpy(&r, &h, 2); return r;
}
__device__ __forceinline__ float bf16_to_f32(bf16_t b) {
    uint16_t h; __builtin_memcpy(&h, &b, 2);
    union U { uint32_t u; float f; };
    U c; c.u = ((uint32_t)h) << 16;
    return c.f;
}

// per-lane async copy: 16 bytes global -> LDS, tracked by ASYNCcnt
__device__ __forceinline__ void async_g2l_b128(uint32_t lds_addr, const void* gaddr) {
    uint64_t ga = (uint64_t)gaddr;
    asm volatile("global_load_async_to_lds_b128 %0, %1, off"
                 :: "v"(lds_addr), "v"(ga) : "memory");
}
__device__ __forceinline__ void wait_asynccnt0() {
    asm volatile("s_wait_asynccnt 0x0" ::: "memory");
}
__device__ __forceinline__ uint32_t lds_offset_of(const bf16_t* p) {
    return (uint32_t)(uintptr_t)(lds_bf16*)p;
}

__device__ __constant__ float GLCM_MEAN[3] = {0.485f, 0.456f, 0.406f};
__device__ __constant__ float GLCM_STD[3]  = {0.229f, 0.224f, 0.225f};

// ------------------------------- utility kernels ---------------------------

__global__ void fill_zero_f32(float* __restrict__ p, int n) {
    int i = blockIdx.x * blockDim.x + threadIdx.x;
    if (i < n) p[i] = 0.0f;
}

__global__ void cvt_f32_to_bf16(const float* __restrict__ s, bf16_t* __restrict__ d, int n) {
    int i = blockIdx.x * blockDim.x + threadIdx.x;
    if (i < n) d[i] = f32_to_bf16(s[i]);
}

// ------------------------------- GLCM --------------------------------------

__global__ void glcm_kernel(const float* __restrict__ x, float* __restrict__ hist) {
    int idx = blockIdx.x * blockDim.x + threadIdx.x;
    const int total = 32 * 3 * 256 * 256;
    if (idx >= total) return;
    int xw = idx & 255;
    int yh = (idx >> 8) & 255;
    int c  = (idx >> 16) % 3;
    int b  = idx / (3 * 256 * 256);
    int base = ((b * 3 + c) * 256 + yh) * 256;
    float mu = GLCM_MEAN[c], sd = GLCM_STD[c];

    auto quant = [&](float v) -> int {
        float xd = fminf(fmaxf(v * sd + mu, 0.0f), 1.0f);
        return (int)(xd * 255.0f);
    };

    int q0 = quant(x[base + xw]);
    float* hbc = hist + (size_t)(b * 3 + c) * 65536;
    if (xw + 1 < 256) {
        int q1 = quant(x[base + xw + 1]);
        atomicAdd(hbc + q0 * 256 + q1, 1.0f);
        atomicAdd(hbc + q1 * 256 + q0, 1.0f);
    }
    if (yh + 1 < 256) {
        int q2 = quant(x[base + 256 + xw]);
        atomicAdd(hbc + q0 * 256 + q2, 1.0f);
        atomicAdd(hbc + q2 * 256 + q0, 1.0f);
    }
}

// ------------------------------- maxpool 2x2 -------------------------------

__global__ void maxpool2_kernel(const bf16_t* __restrict__ in, bf16_t* __restrict__ out,
                                int BC, int H, int W) {
    int Ho = H >> 1, Wo = W >> 1;
    int i = blockIdx.x * blockDim.x + threadIdx.x;
    int total = BC * Ho * Wo;
    if (i >= total) return;
    int xo = i % Wo;
    int yo = (i / Wo) % Ho;
    int bc = i / (Wo * Ho);
    const bf16_t* p = in + ((size_t)bc * H + yo * 2) * W + xo * 2;
    float v0 = bf16_to_f32(p[0]),  v1 = bf16_to_f32(p[1]);
    float v2 = bf16_to_f32(p[W]),  v3 = bf16_to_f32(p[W + 1]);
    out[i] = f32_to_bf16(fmaxf(fmaxf(v0, v1), fmaxf(v2, v3)));
}

// ------------------------- conv as implicit-GEMM WMMA ----------------------
// Block = 128 threads (4 wave32): 32 Cout x 64 pixels of one output row.
// GEMM-K = CIN*KS*KS padded to 32.  Per K-step: weight tile [32x32] and
// im2col tile [32x64] staged to LDS via global_load_async_to_lds_b128 when
// possible (ASYNCcnt path), else scalar gather; then each wave issues two
// v_wmma_f32_16x16x32_bf16 (two Cout tiles share one B fragment).
template <int CIN, int COUT, int KS, int PAD, bool RELU>
__global__ void conv_wmma_kernel(const bf16_t* __restrict__ in,   // [B,CIN,H,W]
                                 const bf16_t* __restrict__ wt,   // [COUT, CIN*KS*KS]
                                 const float*  __restrict__ bias, // [COUT]
                                 bf16_t* __restrict__ out,        // [B,COUT,H,W]
                                 int H, int W) {
    constexpr int CK = CIN * KS * KS;
    constexpr int KSTEPS = (CK + 31) / 32;
    constexpr bool ASYNC_OK = (CK % 32) == 0;   // no K-tail: every staged k valid

    __shared__ bf16_t lds_w[32 * 32];   // [m][k]
    __shared__ bf16_t lds_b[32 * 64];   // [k][px]

    const int segs = W >> 6;
    const int y    = blockIdx.x / segs;
    const int x0   = (blockIdx.x % segs) << 6;
    const int co0  = blockIdx.y << 5;   // 32 output channels per block
    const int b    = blockIdx.z;
    const int tid  = threadIdx.x;
    const int wave = tid >> 5;
    const int lane = tid & 31;
    const int half = lane >> 4;
    const int mn   = lane & 15;

    const uint32_t lw_base = lds_offset_of(lds_w);
    const uint32_t lb_base = lds_offset_of(lds_b);

    // whole 64px segment + halo inside the image? (uniform across block)
    const bool interior = ASYNC_OK && (x0 >= PAD) && (x0 + 64 + PAD <= W) &&
                          (y >= PAD) && (y + PAD < H);

    v8f acc0 = {}, acc1 = {};

    for (int ks = 0; ks < KSTEPS; ++ks) {
        const int k0 = ks * 32;

        // ---- stage weights [32 x 32]: rows are 64B contiguous in global
        if (ASYNC_OK) {
            int row = tid >> 2;          // 0..31
            int seg = tid & 3;           // 4 x 16B per row
            const bf16_t* gp = wt + (size_t)(co0 + row) * CK + k0 + seg * 8;
            async_g2l_b128(lw_base + row * 64 + seg * 16, gp);
        } else {
            if (ks + 1 < KSTEPS)
                __builtin_prefetch(wt + (size_t)(co0 + (tid & 31)) * CK + k0 + 32, 0, 0);
            int e = tid * 8;
            int m = e >> 5;
            int k = e & 31;
            const bf16_t* wp = wt + (size_t)(co0 + m) * CK + (k0 + k);
#pragma unroll
            for (int j = 0; j < 8; ++j) {
                int gk = k0 + k + j;
                lds_w[m * 32 + k + j] = (gk < CK) ? wp[j] : f32_to_bf16(0.0f);
            }
        }

        // ---- stage im2col B tile [32 k x 64 px]
        if (interior) {
            // each k-row is one contiguous 128B global segment: async copy
#pragma unroll
            for (int t = 0; t < 2; ++t) {
                int s    = tid + t * 128;   // 256 x 16B segments
                int k    = s >> 3;
                int part = s & 7;
                int gk   = k0 + k;
                int ci   = gk / (KS * KS);
                int rem  = gk % (KS * KS);
                int ky   = rem / KS;
                int kx   = rem % KS;
                const bf16_t* gp = in + ((size_t)(b * CIN + ci) * H + (y + ky - PAD)) * W
                                      + x0 + kx - PAD + part * 8;
                async_g2l_b128(lb_base + k * 128 + part * 16, gp);
            }
        } else {
            int k   = tid >> 2;
            int pxb = (tid & 3) << 4;
            int gk  = k0 + k;
            int ci  = gk / (KS * KS);
            int rem = gk % (KS * KS);
            int ky  = rem / KS;
            int kx  = rem % KS;
            int iy  = y + ky - PAD;
            bool rowok = (gk < CK) && (iy >= 0) && (iy < H);
            const bf16_t* ip = in + ((size_t)(b * CIN + ci) * H + iy) * W;
#pragma unroll
            for (int j = 0; j < 16; ++j) {
                int ix = x0 + pxb + j + kx - PAD;
                bf16_t v = (rowok && ix >= 0 && ix < W) ? ip[ix] : f32_to_bf16(0.0f);
                lds_b[k * 64 + pxb + j] = v;
            }
        }

        if (ASYNC_OK) wait_asynccnt0();   // drain this wave's async copies
        __syncthreads();                  // make all waves' LDS writes visible

        // ---- gather wave32 WMMA fragments from LDS (ISA 16-bit A/B layout)
        v16bf a0, a1, bfrag;
#pragma unroll
        for (int i = 0; i < 8; ++i) {
            int kk = ((i & 3) << 1) + (half << 3) + ((i >> 2) << 4);
            a0[2 * i]     = lds_w[mn * 32 + kk];
            a0[2 * i + 1] = lds_w[mn * 32 + kk + 1];
            a1[2 * i]     = lds_w[(16 + mn) * 32 + kk];
            a1[2 * i + 1] = lds_w[(16 + mn) * 32 + kk + 1];
            bfrag[2 * i]     = lds_b[kk * 64 + (wave << 4) + mn];
            bfrag[2 * i + 1] = lds_b[(kk + 1) * 64 + (wave << 4) + mn];
        }
        acc0 = __builtin_amdgcn_wmma_f32_16x16x32_bf16(false, a0, false, bfrag,
                                                       (short)0, acc0, false, false);
        acc1 = __builtin_amdgcn_wmma_f32_16x16x32_bf16(false, a1, false, bfrag,
                                                       (short)0, acc1, false, false);
        __syncthreads();
    }

    // ---- epilogue: C/D layout VGPR r -> M = r + 8*half, N = lane&15
    const int n_px = x0 + (wave << 4) + mn;
#pragma unroll
    for (int r = 0; r < 8; ++r) {
        int m = r + (half << 3);
        float v0 = acc0[r] + bias[co0 + m];
        float v1 = acc1[r] + bias[co0 + 16 + m];
        if (RELU) { v0 = fmaxf(v0, 0.0f); v1 = fmaxf(v1, 0.0f); }
        out[((size_t)(b * COUT + co0 + m) * H + y) * W + n_px]      = f32_to_bf16(v0);
        out[((size_t)(b * COUT + co0 + 16 + m) * H + y) * W + n_px] = f32_to_bf16(v1);
    }
}

// ------------------------------- FC GEMM -----------------------------------
// C[M=32, N] = A[M,K] * W[N,K]^T.  One wave per 16x16 tile; per-lane fragment
// gathers are contiguous 16B runs -> compiler emits global_load_b128 pairs.
__global__ void fc_wmma_kernel(const bf16_t* __restrict__ A,   // [32, K]
                               const bf16_t* __restrict__ Wt,  // [N, K]
                               const float*  __restrict__ bias,// [N]
                               bf16_t* __restrict__ out,       // [32, N]
                               int N, int K, int relu) {
    const int n0   = blockIdx.x << 4;
    const int m0   = blockIdx.y << 4;
    const int lane = threadIdx.x & 31;
    const int half = lane >> 4;
    const int mn   = lane & 15;

    v8f acc = {};
    const bf16_t* arow = A  + (size_t)(m0 + mn) * K;
    const bf16_t* brow = Wt + (size_t)(n0 + mn) * K;

    for (int k0 = 0; k0 < K; k0 += 32) {
        v16bf afrag, bfrag;
#pragma unroll
        for (int i = 0; i < 8; ++i) {
            int kk = k0 + ((i & 3) << 1) + (half << 3) + ((i >> 2) << 4);
            afrag[2 * i]     = arow[kk];
            afrag[2 * i + 1] = arow[kk + 1];
            bfrag[2 * i]     = brow[kk];
            bfrag[2 * i + 1] = brow[kk + 1];
        }
        acc = __builtin_amdgcn_wmma_f32_16x16x32_bf16(false, afrag, false, bfrag,
                                                      (short)0, acc, false, false);
    }
#pragma unroll
    for (int r = 0; r < 8; ++r) {
        int m = r + (half << 3);
        float v = acc[r] + bias[n0 + mn];
        if (relu) v = fmaxf(v, 0.0f);
        out[(size_t)(m0 + m) * N + n0 + mn] = f32_to_bf16(v);
    }
}

// ------------------------------- FC3 + sigmoid -----------------------------

__global__ void fc3_sigmoid_kernel(const bf16_t* __restrict__ h,  // [32,256]
                                   const float* __restrict__ w,   // [256]
                                   const float* __restrict__ b,   // [1]
                                   float* __restrict__ out) {     // [32]
    int i = threadIdx.x;  // batch row, 32 threads
    float s = 0.0f;
    for (int k = 0; k < 256; ++k) s += bf16_to_f32(h[i * 256 + k]) * w[k];
    s += b[0];
    out[i] = 1.0f / (1.0f + __expf(-s));
}

// ------------------------------- launch ------------------------------------

extern "C" void kernel_launch(void* const* d_in, const int* in_sizes, int n_in,
                              void* d_out, int out_size, void* d_ws, size_t ws_size,
                              hipStream_t stream) {
    (void)in_sizes; (void)n_in; (void)out_size; (void)ws_size;

    const float* x   = (const float*)d_in[0];
    const float* w1  = (const float*)d_in[1];
    const float* b1  = (const float*)d_in[2];
    const float* w2  = (const float*)d_in[3];
    const float* b2  = (const float*)d_in[4];
    const float* w3  = (const float*)d_in[5];
    const float* b3  = (const float*)d_in[6];
    const float* w4  = (const float*)d_in[7];
    const float* b4  = (const float*)d_in[8];
    const float* w5  = (const float*)d_in[9];
    const float* b5  = (const float*)d_in[10];
    const float* w6  = (const float*)d_in[11];
    const float* b6  = (const float*)d_in[12];
    const float* fw1 = (const float*)d_in[13];
    const float* fb1 = (const float*)d_in[14];
    const float* fw2 = (const float*)d_in[15];
    const float* fb2 = (const float*)d_in[16];
    const float* fw3 = (const float*)d_in[17];
    const float* fb3 = (const float*)d_in[18];

    char* ws = (char*)d_ws;
    size_t off = 0;
    auto alloc = [&](size_t bytes) -> char* {
        char* p = ws + off;
        off = (off + bytes + 255) & ~(size_t)255;
        return p;
    };

    // persistent bf16 weights
    bf16_t* w1b  = (bf16_t*)alloc(864 * 2);
    bf16_t* w2b  = (bf16_t*)alloc(25600 * 2);
    bf16_t* w3b  = (bf16_t*)alloc(18432 * 2);
    bf16_t* w4b  = (bf16_t*)alloc(102400 * 2);
    bf16_t* w5b  = (bf16_t*)alloc(73728 * 2);
    bf16_t* w6b  = (bf16_t*)alloc(409600 * 2);
    bf16_t* fw1b = (bf16_t*)alloc((size_t)33554432 * 2);
    bf16_t* fw2b = (bf16_t*)alloc(65536 * 2);
    // ping-pong activation slots
    char* slot1 = alloc((size_t)134217728);   // up to 32x32x256x256 bf16
    char* slot2 = alloc((size_t)134217728);
    char* slot3 = alloc((size_t)33554432);    // up to 32x32x128x128 bf16

    auto cvt = [&](const float* s, bf16_t* d, int n) {
        cvt_f32_to_bf16<<<(n + 255) / 256, 256, 0, stream>>>(s, d, n);
    };
    cvt(w1, w1b, 864);
    cvt(w2, w2b, 25600);
    cvt(w3, w3b, 18432);
    cvt(w4, w4b, 102400);
    cvt(w5, w5b, 73728);
    cvt(w6, w6b, 409600);
    cvt(fw1, fw1b, 33554432);
    cvt(fw2, fw2b, 65536);

    // GLCM histogram (f32 in slot2), then convert to bf16 act0 in slot3
    float* hist = (float*)slot2;
    const int histn = 96 * 65536;
    fill_zero_f32<<<(histn + 255) / 256, 256, 0, stream>>>(hist, histn);
    glcm_kernel<<<(32 * 3 * 65536 + 255) / 256, 256, 0, stream>>>(x, hist);
    cvt_f32_to_bf16<<<(histn + 255) / 256, 256, 0, stream>>>(hist, (bf16_t*)slot3, histn);

    // conv1 (relu): slot3 -> slot1          [32,32,256,256]
    conv_wmma_kernel<3, 32, 3, 1, true>
        <<<dim3(4 * 256, 1, 32), 128, 0, stream>>>((bf16_t*)slot3, w1b, b1, (bf16_t*)slot1, 256, 256);
    // conv2: slot1 -> slot2                 [32,32,256,256]
    conv_wmma_kernel<32, 32, 5, 2, false>
        <<<dim3(4 * 256, 1, 32), 128, 0, stream>>>((bf16_t*)slot1, w2b, b2, (bf16_t*)slot2, 256, 256);
    // pool1: slot2 -> slot3                 [32,32,128,128]
    {
        int n = 32 * 32 * 128 * 128;
        maxpool2_kernel<<<(n + 255) / 256, 256, 0, stream>>>((bf16_t*)slot2, (bf16_t*)slot3, 32 * 32, 256, 256);
    }
    // conv3 (relu): slot3 -> slot1          [32,64,128,128]
    conv_wmma_kernel<32, 64, 3, 1, true>
        <<<dim3(2 * 128, 2, 32), 128, 0, stream>>>((bf16_t*)slot3, w3b, b3, (bf16_t*)slot1, 128, 128);
    // conv4: slot1 -> slot2                 [32,64,128,128]
    conv_wmma_kernel<64, 64, 5, 2, false>
        <<<dim3(2 * 128, 2, 32), 128, 0, stream>>>((bf16_t*)slot1, w4b, b4, (bf16_t*)slot2, 128, 128);
    // pool2: slot2 -> slot3                 [32,64,64,64]
    {
        int n = 32 * 64 * 64 * 64;
        maxpool2_kernel<<<(n + 255) / 256, 256, 0, stream>>>((bf16_t*)slot2, (bf16_t*)slot3, 32 * 64, 128, 128);
    }
    // conv5 (relu): slot3 -> slot1          [32,128,64,64]
    conv_wmma_kernel<64, 128, 3, 1, true>
        <<<dim3(1 * 64, 4, 32), 128, 0, stream>>>((bf16_t*)slot3, w5b, b5, (bf16_t*)slot1, 64, 64);
    // conv6: slot1 -> slot2                 [32,128,64,64]
    conv_wmma_kernel<128, 128, 5, 2, false>
        <<<dim3(1 * 64, 4, 32), 128, 0, stream>>>((bf16_t*)slot1, w6b, b6, (bf16_t*)slot2, 64, 64);
    // pool3: slot2 -> slot3                 [32,128,32,32] == FC1 input [32,131072]
    {
        int n = 32 * 128 * 32 * 32;
        maxpool2_kernel<<<(n + 255) / 256, 256, 0, stream>>>((bf16_t*)slot2, (bf16_t*)slot3, 32 * 128, 64, 64);
    }

    // FC1 (relu): slot3 -> slot1 [32,256]
    fc_wmma_kernel<<<dim3(16, 2), 32, 0, stream>>>((bf16_t*)slot3, fw1b, fb1, (bf16_t*)slot1, 256, 131072, 1);
    // FC2 (relu): slot1 -> slot2 [32,256]
    fc_wmma_kernel<<<dim3(16, 2), 32, 0, stream>>>((bf16_t*)slot1, fw2b, fb2, (bf16_t*)slot2, 256, 256, 1);
    // FC3 + sigmoid -> d_out [32]
    fc3_sigmoid_kernel<<<1, 32, 0, stream>>>((bf16_t*)slot2, fw3, fb3, (float*)d_out);
}
